// ResidualMamba_32607391711721
// MI455X (gfx1250) — matrix operand order repaired and checked by reading, not compiled, not run on hardware
//
#include <hip/hip_runtime.h>
#include <hip/hip_bf16.h>
#include <math.h>

// ---------------------------------------------------------------------------
// Shapes (compile-time constants from the reference)
// ---------------------------------------------------------------------------
#define D_MODEL 768
#define D_STATE 32
#define D_CONV 4
#define D_INNER 1536
#define DT_RANK 48
#define MLP_HID 3072
#define BATCH 2
#define SEQ 1024
#define MROWS (BATCH * SEQ)              // 2048 token rows
#define XDBL_W (DT_RANK + 2 * D_STATE)   // 112

// ---------------------------------------------------------------------------
// WMMA / TDM types (CDNA5 wave32)
// ---------------------------------------------------------------------------
typedef __attribute__((ext_vector_type(16))) __bf16 bf16x16;
typedef __attribute__((ext_vector_type(8)))  float  floatx8;
typedef __attribute__((ext_vector_type(4)))  unsigned int u32x4;
typedef __attribute__((ext_vector_type(8)))  int          i32x8;
typedef __attribute__((ext_vector_type(4)))  int          i32x4;

union Frag16 {
    bf16x16 v;
    unsigned int u[8];
};

__device__ __forceinline__ unsigned short f32_to_bf16(float f) {
    unsigned int u = __float_as_uint(f);
    unsigned int r = u + 0x7FFFu + ((u >> 16) & 1u);   // round-to-nearest-even
    return (unsigned short)(r >> 16);
}

__device__ __forceinline__ float sigmoidf_dev(float x) {
    return 1.0f / (1.0f + __expf(-x));
}

// LDS byte offset of a generic pointer to __shared__ (aperture puts the LDS
// offset in the low 32 bits of the generic address).
__device__ __forceinline__ unsigned lds_offset_of(const void* p) {
    return (unsigned)(unsigned long long)(uintptr_t)p;
}

// ---------------------------------------------------------------------------
// TDM: DMA a 2D bf16 tile (tile_k x tile_rows) from global into LDS, with
// 1-DWORD padding every 16 DWORDs (=> LDS row stride of 34 halfs for a
// 32-half tile row).  OOB (beyond rem_k / rem_rows) zero-fills.
// ---------------------------------------------------------------------------
__device__ __forceinline__ void tdm_load_tile_bf16(
    unsigned lds_byte_addr,
    const unsigned short* gptr,      // global tile start
    unsigned rem_k,                  // valid elements along dim0 from tile start
    unsigned rem_rows,               // valid rows along dim1 from tile start
    unsigned stride_elems,           // row stride in elements
    unsigned tile_k,                 // tile_dim0 (elements)
    unsigned tile_rows)              // tile_dim1 (rows)
{
    unsigned long long ga = (unsigned long long)(uintptr_t)gptr;
    u32x4 g0;
    g0[0] = 1u;                                                // count=1, user D#
    g0[1] = lds_byte_addr;                                     // lds_addr
    g0[2] = (unsigned)(ga & 0xFFFFFFFFu);                      // global_addr lo
    g0[3] = (unsigned)((ga >> 32) & 0x01FFFFFFu) | (2u << 30); // addr hi | type=2

    unsigned long long s0 = (unsigned long long)stride_elems;
    i32x8 g1;
    g1[0] = (int)((1u << 16)      // data_size = 1 -> 2 bytes
                | (1u << 20)      // pad_enable
                | (3u << 22)      // pad_interval code 3 -> every 16 DWORDs
                | (0u << 25));    // pad_amount 0 -> 1 DWORD
    g1[1] = (int)((rem_k & 0xFFFFu) << 16);                        // dim0 lo16
    g1[2] = (int)((rem_k >> 16) | ((rem_rows & 0xFFFFu) << 16));   // dim0 hi | dim1 lo
    g1[3] = (int)((rem_rows >> 16) | (tile_k << 16));              // dim1 hi | tile_dim0
    g1[4] = (int)(tile_rows);                                      // tile_dim1 (tile_dim2=0)
    g1[5] = (int)(s0 & 0xFFFFFFFFu);                               // stride0 lo32
    g1[6] = (int)((s0 >> 32) & 0xFFFFu);                           // stride0 hi16 (stride1=0)
    g1[7] = 0;
    i32x4 z4 = {0, 0, 0, 0};                 // 2D tensor: groups 2/3 unused
    i32x8 z8 = {0, 0, 0, 0, 0, 0, 0, 0};     // 6-arg form: extra group, unused
    __builtin_amdgcn_tensor_load_to_lds(g0, g1, z4, z4, z8, 0);
}

// ---------------------------------------------------------------------------
// Kernel 1: fused double LayerNorm -> bf16 outputs (GEMM A operands).
// ---------------------------------------------------------------------------
__global__ __launch_bounds__(256) void ln2_kernel(
    const float* __restrict__ x,
    const float* __restrict__ w1, const float* __restrict__ b1,
    const float* __restrict__ w2, const float* __restrict__ b2,
    unsigned short* __restrict__ xn1, unsigned short* __restrict__ xn2)
{
    const int row = blockIdx.x;
    const float* xr = x + (size_t)row * D_MODEL;
    float s = 0.f, s2 = 0.f;
    for (int i = threadIdx.x; i < D_MODEL; i += 256) {
        float v = xr[i];
        s += v; s2 += v * v;
    }
    for (int o = 16; o >= 1; o >>= 1) {
        s  += __shfl_xor(s,  o, 32);
        s2 += __shfl_xor(s2, o, 32);
    }
    __shared__ float red[18];
    const int lane = threadIdx.x & 31;
    const int wv   = threadIdx.x >> 5;
    if (lane == 0) { red[wv] = s; red[8 + wv] = s2; }
    __syncthreads();
    if (threadIdx.x == 0) {
        float ts = 0.f, ts2 = 0.f;
        for (int w = 0; w < 8; ++w) { ts += red[w]; ts2 += red[8 + w]; }
        float mu  = ts * (1.0f / D_MODEL);
        float var = ts2 * (1.0f / D_MODEL) - mu * mu;
        red[16] = mu;
        red[17] = rsqrtf(var + 1e-5f);
    }
    __syncthreads();
    const float mu = red[16], rs = red[17];
    for (int i = threadIdx.x; i < D_MODEL; i += 256) {
        float v = (xr[i] - mu) * rs;
        xn1[(size_t)row * D_MODEL + i] = f32_to_bf16(v * w1[i] + b1[i]);
        xn2[(size_t)row * D_MODEL + i] = f32_to_bf16(v * w2[i] + b2[i]);
    }
}

// ---------------------------------------------------------------------------
// Kernel 2: f32 -> bf16 convert (weight prep).
// ---------------------------------------------------------------------------
__global__ __launch_bounds__(256) void cvt_bf16_kernel(
    const float* __restrict__ src, unsigned short* __restrict__ dst, int n)
{
    const int i = blockIdx.x * 256 + threadIdx.x;
    if (i < n) dst[i] = f32_to_bf16(src[i]);
}

// ---------------------------------------------------------------------------
// Kernel 3: TDM-staged bf16 WMMA GEMM:
//   C[M,N] (f32, optional) / Cbf[M,N] (bf16, optional) = A[M,lda] * W[N,K]^T
//   epi: 0=raw, 1=+bias & exact GELU, 3=+bias+add0+add1 (fused residual)
// Block tile 128x128, K-step 32, 8 waves x (2x4) V_WMMA_F32_16X16X32_BF16,
// double-buffered LDS fed by TENSOR_LOAD_TO_LDS (wave 0 issues, TENSORcnt).
// ---------------------------------------------------------------------------
#define BM 128
#define BN 128
#define BK 32
#define LDSW 34                 // padded LDS row stride in halfs
#define A_TILE (BM * LDSW)      // halfs
#define B_TILE (BN * LDSW)      // halfs

__global__ __launch_bounds__(256) void gemm_bf16_wmma_tdm_kernel(
    const unsigned short* __restrict__ A, int lda,
    const unsigned short* __restrict__ W,
    float* __restrict__ C, unsigned short* __restrict__ Cbf,
    int M, int N, int K, int epi,
    const float* __restrict__ bias,
    const float* __restrict__ add0,
    const float* __restrict__ add1)
{
    __shared__ __align__(16) unsigned short smem[2 * (A_TILE + B_TILE)];

    const int tid  = threadIdx.x;
    const int lane = tid & 31;
    const int wave = tid >> 5;          // 0..7
    const int wm   = wave & 3;          // 4 waves along M (32 rows each)
    const int wn   = wave >> 2;         // 2 waves along N (64 cols each)
    const int m0   = blockIdx.x * BM;
    const int n0   = blockIdx.y * BN;
    const int nk   = (K + BK - 1) / BK;

    const unsigned lbase = lds_offset_of(smem);
    const unsigned bufBytes = (A_TILE + B_TILE) * 2u;

    floatx8 acc[2][4];
    #pragma unroll
    for (int i = 0; i < 2; ++i)
        #pragma unroll
        for (int j = 0; j < 4; ++j)
            acc[i][j] = (floatx8)0.0f;

    // prologue: DMA tile 0 into buffer 0
    if (wave == 0) {
        tdm_load_tile_bf16(lbase,
                           A + (size_t)m0 * lda, (unsigned)K, (unsigned)(M - m0),
                           (unsigned)lda, BK, BM);
        tdm_load_tile_bf16(lbase + A_TILE * 2u,
                           W + (size_t)n0 * K, (unsigned)K, (unsigned)(N - n0),
                           (unsigned)K, BK, BN);
    }

    for (int kt = 0; kt < nk; ++kt) {
        const int cur = kt & 1;
        if (wave == 0) {
            __builtin_amdgcn_s_wait_tensorcnt(0);   // current buffer landed
        }
        __syncthreads();
        if (wave == 0 && (kt + 1) < nk) {           // prefetch next tile
            const int k0n = (kt + 1) * BK;
            const unsigned nb = lbase + (unsigned)(cur ^ 1) * bufBytes;
            tdm_load_tile_bf16(nb,
                               A + (size_t)m0 * lda + k0n,
                               (unsigned)(K - k0n), (unsigned)(M - m0),
                               (unsigned)lda, BK, BM);
            tdm_load_tile_bf16(nb + A_TILE * 2u,
                               W + (size_t)n0 * K + k0n,
                               (unsigned)(K - k0n), (unsigned)(N - n0),
                               (unsigned)K, BK, BN);
        }

        const unsigned short* As = smem + (size_t)cur * (A_TILE + B_TILE);
        const unsigned short* Bs = As + A_TILE;

        // fragment loads per documented CDNA5 16-bit VGPR layouts
        const int mr = lane & 15;
        const int hh = lane >> 4;
        Frag16 af[2], bf[4];
        #pragma unroll
        for (int ti = 0; ti < 2; ++ti) {
            const unsigned short* base = &As[(wm * 32 + ti * 16 + mr) * LDSW];
            #pragma unroll
            for (int v = 0; v < 8; ++v) {
                // A 16x32: lanes0-15 K=0..7,16..23 ; lanes16-31 K=8..15,24..31
                const int kk = ((v & 4) ? 16 : 0) + hh * 8 + (v & 3) * 2;
                af[ti].u[v] = *(const unsigned int*)(base + kk);
            }
        }
        #pragma unroll
        for (int tj = 0; tj < 4; ++tj) {
            const unsigned short* base = &Bs[(wn * 64 + tj * 16 + mr) * LDSW];
            #pragma unroll
            for (int v = 0; v < 8; ++v) {
                // B 32x16: lanes0-15 K=0..7,16..23 ; lanes16-31 K=8..15,24..31
                const int kk = ((v & 4) ? 16 : 0) + hh * 8 + (v & 3) * 2;
                bf[tj].u[v] = *(const unsigned int*)(base + kk);
            }
        }
        #pragma unroll
        for (int ti = 0; ti < 2; ++ti)
            #pragma unroll
            for (int tj = 0; tj < 4; ++tj)
                acc[ti][tj] = __builtin_amdgcn_wmma_f32_16x16x32_bf16(
                    false, af[ti].v, false, bf[tj].v,
                    (short)0, acc[ti][tj], false, false);
        __syncthreads();
    }

    // epilogue + store (C/D layout: N = lane&15, M = r + (lane>>4)*8)
    const int nl = lane & 15;
    const int mh = lane >> 4;
    #pragma unroll
    for (int ti = 0; ti < 2; ++ti) {
        #pragma unroll
        for (int tj = 0; tj < 4; ++tj) {
            const int gcol = n0 + wn * 64 + tj * 16 + nl;
            if (gcol >= N) continue;
            const float bv = bias ? bias[gcol] : 0.f;
            #pragma unroll
            for (int r = 0; r < 8; ++r) {
                const int grow = m0 + wm * 32 + ti * 16 + mh * 8 + r;
                const size_t idx = (size_t)grow * N + gcol;
                float v = acc[ti][tj][r];
                if (epi == 1) {
                    float t = v + bv;
                    v = 0.5f * t * (1.0f + erff(t * 0.70710678118654752f));
                } else if (epi == 3) {
                    v = v + bv + add0[idx] + add1[idx];
                }
                if (C)   C[idx]   = v;
                if (Cbf) Cbf[idx] = f32_to_bf16(v);
            }
        }
    }
}

// ---------------------------------------------------------------------------
// Kernel 4: causal depthwise conv1d (D_CONV=4) + bias + SiLU.
// Writes f32 (for the scan) and bf16 (A operand of x_proj GEMM).
// ---------------------------------------------------------------------------
__global__ __launch_bounds__(256) void conv_silu_kernel(
    const float* __restrict__ xz,
    const float* __restrict__ cw, const float* __restrict__ cb,
    float* __restrict__ xconv_f, unsigned short* __restrict__ xconv_bf)
{
    const int idx = blockIdx.x * 256 + threadIdx.x;   // over B*L*D_INNER
    if (idx >= MROWS * D_INNER) return;
    const int d  = idx % D_INNER;
    const int bl = idx / D_INNER;
    const int l  = bl % SEQ;
    float acc = cb[d];
    #pragma unroll
    for (int t = 0; t < D_CONV; ++t) {
        const int ll = l - (D_CONV - 1) + t;
        if (ll >= 0)
            acc += cw[d * D_CONV + t] *
                   xz[(size_t)(bl - (D_CONV - 1 - t)) * (2 * D_INNER) + d];
    }
    const float v = acc * sigmoidf_dev(acc);
    xconv_f[idx]  = v;
    xconv_bf[idx] = f32_to_bf16(v);
}

// ---------------------------------------------------------------------------
// Kernel 5: selective scan.  One wave32 per (b,d) channel; lane = state n.
// ---------------------------------------------------------------------------
__global__ __launch_bounds__(256) void scan_kernel(
    const float* __restrict__ dtp,
    const float* __restrict__ dtb,
    const float* __restrict__ xconv,
    const float* __restrict__ xdbl,
    const float* __restrict__ xz,
    const float* __restrict__ A_log,
    const float* __restrict__ Dp,
    unsigned short* __restrict__ ybuf_bf)
{
    const int wave = blockIdx.x * 8 + (threadIdx.x >> 5);
    if (wave >= BATCH * D_INNER) return;
    const int b    = wave / D_INNER;
    const int d    = wave % D_INNER;
    const int lane = threadIdx.x & 31;

    const float An    = -__expf(A_log[d * D_STATE + lane]);
    const float Dd    = Dp[d];
    const float dbias = dtb[d];
    float h = 0.f;

    for (int l = 0; l < SEQ; ++l) {
        const int bl = b * SEQ + l;
        float draw = 0.f, xm = 0.f, z = 0.f;
        if (lane == 0) {
            draw = dtp[(size_t)bl * D_INNER + d] + dbias;
            xm   = xconv[(size_t)bl * D_INNER + d];
            z    = xz[(size_t)bl * (2 * D_INNER) + D_INNER + d];
        }
        draw = __shfl(draw, 0, 32);
        xm   = __shfl(xm,   0, 32);
        z    = __shfl(z,    0, 32);
        const float delta = (draw > 20.f) ? draw : log1pf(__expf(draw));
        const float Bn = xdbl[(size_t)bl * XDBL_W + DT_RANK + lane];
        const float Cn = xdbl[(size_t)bl * XDBL_W + DT_RANK + D_STATE + lane];
        h = __expf(delta * An) * h + (delta * Bn) * xm;
        float c = h * Cn;
        #pragma unroll
        for (int o = 16; o >= 1; o >>= 1) c += __shfl_xor(c, o, 32);
        if (lane == 0) {
            const float y = c + Dd * xm;
            ybuf_bf[(size_t)bl * D_INNER + d] = f32_to_bf16(y * (z * sigmoidf_dev(z)));
        }
    }
}

// ---------------------------------------------------------------------------
// Launch
// ---------------------------------------------------------------------------
extern "C" void kernel_launch(void* const* d_in, const int* in_sizes, int n_in,
                              void* d_out, int out_size, void* d_ws, size_t ws_size,
                              hipStream_t stream)
{
    (void)in_sizes; (void)n_in; (void)out_size; (void)ws_size;
    const float* x         = (const float*)d_in[0];
    const float* norm1_w   = (const float*)d_in[1];
    const float* norm1_b   = (const float*)d_in[2];
    const float* norm2_w   = (const float*)d_in[3];
    const float* norm2_b   = (const float*)d_in[4];
    const float* in_proj_w = (const float*)d_in[5];
    const float* conv_w    = (const float*)d_in[6];
    const float* conv_b    = (const float*)d_in[7];
    const float* x_proj_w  = (const float*)d_in[8];
    const float* dt_proj_w = (const float*)d_in[9];
    const float* dt_proj_b = (const float*)d_in[10];
    const float* A_log     = (const float*)d_in[11];
    const float* Dp        = (const float*)d_in[12];
    const float* out_proj_w= (const float*)d_in[13];
    const float* fc1_w     = (const float*)d_in[14];
    const float* fc1_b     = (const float*)d_in[15];
    const float* fc2_w     = (const float*)d_in[16];
    const float* fc2_b     = (const float*)d_in[17];
    float* out = (float*)d_out;

    // workspace carve-up (byte allocator, 256B aligned)
    char* wsb = (char*)d_ws;
    size_t off = 0;
    auto carve = [&](size_t bytes) -> char* {
        char* p = wsb + off;
        off += (bytes + 255) & ~(size_t)255;
        return p;
    };
    unsigned short* xn1_bf   = (unsigned short*)carve((size_t)MROWS * D_MODEL * 2);
    unsigned short* xn2_bf   = (unsigned short*)carve((size_t)MROWS * D_MODEL * 2);
    float*          xz       = (float*)carve((size_t)MROWS * 2 * D_INNER * 4);
    float*          xconv_f  = (float*)carve((size_t)MROWS * D_INNER * 4);
    unsigned short* xconv_bf = (unsigned short*)carve((size_t)MROWS * D_INNER * 2);
    float*          xdbl_f   = (float*)carve((size_t)MROWS * XDBL_W * 4);
    unsigned short* xdbl_bf  = (unsigned short*)carve((size_t)MROWS * XDBL_W * 2);
    float*          dtp      = (float*)carve((size_t)MROWS * D_INNER * 4);
    unsigned short* ybuf_bf  = (unsigned short*)carve((size_t)MROWS * D_INNER * 2);
    float*          mamba    = (float*)carve((size_t)MROWS * D_MODEL * 4);
    unsigned short* h1_bf    = (unsigned short*)carve((size_t)MROWS * MLP_HID * 2);
    unsigned short* w_inp_bf = (unsigned short*)carve((size_t)2 * D_INNER * D_MODEL * 2);
    unsigned short* w_xp_bf  = (unsigned short*)carve((size_t)XDBL_W * D_INNER * 2);
    unsigned short* w_dt_bf  = (unsigned short*)carve((size_t)D_INNER * DT_RANK * 2);
    unsigned short* w_out_bf = (unsigned short*)carve((size_t)D_MODEL * D_INNER * 2);
    unsigned short* w_fc1_bf = (unsigned short*)carve((size_t)MLP_HID * D_MODEL * 2);
    unsigned short* w_fc2_bf = (unsigned short*)carve((size_t)D_MODEL * MLP_HID * 2);

    // 0. weight conversion to bf16
    auto cvt = [&](const float* s, unsigned short* d, int n) {
        cvt_bf16_kernel<<<(n + 255) / 256, 256, 0, stream>>>(s, d, n);
    };
    cvt(in_proj_w,  w_inp_bf, 2 * D_INNER * D_MODEL);
    cvt(x_proj_w,   w_xp_bf,  XDBL_W * D_INNER);
    cvt(dt_proj_w,  w_dt_bf,  D_INNER * DT_RANK);
    cvt(out_proj_w, w_out_bf, D_MODEL * D_INNER);
    cvt(fc1_w,      w_fc1_bf, MLP_HID * D_MODEL);
    cvt(fc2_w,      w_fc2_bf, D_MODEL * MLP_HID);

    // 1. double LayerNorm -> bf16
    ln2_kernel<<<MROWS, 256, 0, stream>>>(x, norm1_w, norm1_b, norm2_w, norm2_b,
                                          xn1_bf, xn2_bf);
    // 2. in_proj: xz = xn1 @ in_proj_w^T  (f32 out)
    gemm_bf16_wmma_tdm_kernel<<<dim3(MROWS / BM, (2 * D_INNER) / BN), 256, 0, stream>>>(
        xn1_bf, D_MODEL, w_inp_bf, xz, nullptr,
        MROWS, 2 * D_INNER, D_MODEL, 0, nullptr, nullptr, nullptr);
    // 3. causal conv + SiLU (f32 + bf16 out)
    conv_silu_kernel<<<(MROWS * D_INNER) / 256, 256, 0, stream>>>(
        xz, conv_w, conv_b, xconv_f, xconv_bf);
    // 4. x_proj: xdbl = xconv @ x_proj_w^T  (f32 for scan + bf16 for dt_proj)
    gemm_bf16_wmma_tdm_kernel<<<dim3(MROWS / BM, 1), 256, 0, stream>>>(
        xconv_bf, D_INNER, w_xp_bf, xdbl_f, xdbl_bf,
        MROWS, XDBL_W, D_INNER, 0, nullptr, nullptr, nullptr);
    // 5. dt_proj: dtp = xdbl[:, :48] @ dt_proj_w^T  (K=48, lda=112; TDM zero-fills)
    gemm_bf16_wmma_tdm_kernel<<<dim3(MROWS / BM, D_INNER / BN), 256, 0, stream>>>(
        xdbl_bf, XDBL_W, w_dt_bf, dtp, nullptr,
        MROWS, D_INNER, DT_RANK, 0, nullptr, nullptr, nullptr);
    // 6. selective scan (+gating) -> bf16
    scan_kernel<<<(BATCH * D_INNER) / 8, 256, 0, stream>>>(
        dtp, dt_proj_b, xconv_f, xdbl_f, xz, A_log, Dp, ybuf_bf);
    // 7. out_proj: mamba = ybuf @ out_proj_w^T  (f32)
    gemm_bf16_wmma_tdm_kernel<<<dim3(MROWS / BM, D_MODEL / BN), 256, 0, stream>>>(
        ybuf_bf, D_INNER, w_out_bf, mamba, nullptr,
        MROWS, D_MODEL, D_INNER, 0, nullptr, nullptr, nullptr);
    // 8. fc1 + bias + exact GELU -> bf16 hidden
    gemm_bf16_wmma_tdm_kernel<<<dim3(MROWS / BM, MLP_HID / BN), 256, 0, stream>>>(
        xn2_bf, D_MODEL, w_fc1_bf, nullptr, h1_bf,
        MROWS, MLP_HID, D_MODEL, 1, fc1_b, nullptr, nullptr);
    // 9. fc2 + bias + x + mamba residual -> out (f32)
    gemm_bf16_wmma_tdm_kernel<<<dim3(MROWS / BM, D_MODEL / BN), 256, 0, stream>>>(
        h1_bf, MLP_HID, w_fc2_bf, out, nullptr,
        MROWS, D_MODEL, MLP_HID, 3, fc2_b, x, mamba);
}
